// HybridQLSTM_65481071398997
// MI455X (gfx1250) — matrix operand order repaired and checked by reading, not compiled, not run on hardware
//
#include <hip/hip_runtime.h>

typedef _Float16 half16 __attribute__((ext_vector_type(16)));
typedef _Float16 half8  __attribute__((ext_vector_type(8)));
typedef float    f32x8  __attribute__((ext_vector_type(8)));

#define T_STEPS 128
#define BATCH   512
#define DIM     512
#define HID     512
#define NC      32
#define NQ      10
#define R_ROWS  (T_STEPS * BATCH)   // 65536 (t,b) rows

// ---------------------------------------------------------------------------
// Kernel 1: quantum-kernel values  k[g][row][c] = prod_i |cos(0.5*(x_i-ctr_i))|
// stored as f16, row-major [4][R_ROWS][32]  (16 MB, L2-resident).
// One block per (t,b) row; 128 threads = 4 gates x 32 centres.
// ---------------------------------------------------------------------------
__global__ __launch_bounds__(128)
void qk_kernel(const float* __restrict__ x,
               const float* __restrict__ centres,
               _Float16* __restrict__ kout) {
  __shared__ float xs[NQ];
  const int row = blockIdx.x;
  const int tid = threadIdx.x;
  if (tid < NQ) xs[tid] = x[(size_t)row * DIM + tid];
  __syncthreads();

  const int g = tid >> 5;
  const int c = tid & 31;
  const float* ctr = centres + (size_t)(g * NC + c) * (DIM + HID);

  float p = 1.0f;
#pragma unroll
  for (int i = 0; i < NQ; ++i) {
    float d = 0.5f * (xs[i] - ctr[i]);
    p *= fabsf(__cosf(d));
  }
  kout[((size_t)g * R_ROWS + row) * NC + c] = (_Float16)p;
}

// ---------------------------------------------------------------------------
// Kernel 2: fused gate-GEMM (WMMA f16 16x16x32, K==NC in one shot) + LSTM
// recurrence. One wave per 16x16 (b,h) tile; c/h live in VGPRs across all T.
// Gate nonlinearities use the hardware v_rcp_f32 / v_exp_f32 fast path
// instead of IEEE division (div_scale + NR refine), shortening the serial
// recurrence critical path.
// ---------------------------------------------------------------------------
__device__ __forceinline__ float fast_sigmoid(float v) {
  // 1 / (1 + e^-v)  via v_exp_f32 + v_rcp_f32
  return __builtin_amdgcn_rcpf(1.0f + __expf(-v));
}
__device__ __forceinline__ float fast_tanh(float v) {
  // 1 - 2 / (e^(2v) + 1)
  return 1.0f - 2.0f * __builtin_amdgcn_rcpf(__expf(2.0f * v) + 1.0f);
}

__global__ __launch_bounds__(128)
void lstm_kernel(const _Float16* __restrict__ kbuf,
                 const float* __restrict__ lin_w,
                 const float* __restrict__ lin_b,
                 float* __restrict__ out) {
  const int lane = threadIdx.x & 31;
  const int wave = threadIdx.x >> 5;
  const int tile = blockIdx.x * 4 + wave;        // 1024 tiles total
  const int bt = tile / (HID / 16);
  const int ht = tile % (HID / 16);
  const int b0 = bt * 16;
  const int h0 = ht * 16;

  const int nIdx   = lane & 15;                  // N / M-row index within tile
  const int hiHalf = lane >> 4;                  // lane group (0 or 1)

  // ---- B operands (weights), t-invariant: lane holds column h0+nIdx,
  // K contiguous in lin_w[g][h][c]. Convert f32 -> f16 once. ----
  half16 wmat[4];
  float  bias[4];
#pragma unroll
  for (int g = 0; g < 4; ++g) {
    const float* wp =
        lin_w + (size_t)(g * HID + h0 + nIdx) * NC + hiHalf * 8;
    half16 wv;
#pragma unroll
    for (int j = 0; j < 16; ++j) wv[j] = (_Float16)wp[j < 8 ? j : j + 8];
    wmat[g] = wv;
    bias[g] = lin_b[g * HID + h0 + nIdx];
  }

  f32x8 c = {};   // cell state tile (8 VGPRs)
  f32x8 h = {};   // hidden state tile

  const int aRow   = lane & 15;                  // A-matrix row for this lane
  const int aKBase = hiHalf * 8;                 // A K-group base (in halves)

  for (int t = 0; t < T_STEPS; ++t) {
    const size_t row = (size_t)t * BATCH + b0 + aRow;

    f32x8 pre[4];
#pragma unroll
    for (int g = 0; g < 4; ++g) {
      const half8* ap =
          (const half8*)(kbuf + ((size_t)g * R_ROWS + row) * NC + aKBase);
      half8 lo = ap[0];                          // K = kb .. kb+7
      half8 hi = ap[2];                          // K = kb+16 .. kb+23
      half16 a = __builtin_shufflevector(lo, hi, 0, 1, 2, 3, 4, 5, 6, 7,
                                         8, 9, 10, 11, 12, 13, 14, 15);
      if (t + 1 < T_STEPS)                       // global_prefetch_b8 of next t
        __builtin_prefetch(
            (const void*)(kbuf + ((size_t)g * R_ROWS + row + BATCH) * NC +
                          aKBase),
            0, 1);
      f32x8 cacc;
#pragma unroll
      for (int j = 0; j < 8; ++j) cacc[j] = bias[g];   // bias depends on N only
      pre[g] = __builtin_amdgcn_wmma_f32_16x16x32_f16(
          false, a, false, wmat[g], (short)0, cacc, false, false);
    }

    // ---- elementwise LSTM update, all in registers ----
#pragma unroll
    for (int j = 0; j < 8; ++j) {
      float fv = fast_sigmoid(pre[0][j]);
      float iv = fast_sigmoid(pre[1][j]);
      float gv = fast_tanh(pre[2][j]);
      float ov = fast_sigmoid(pre[3][j]);
      float cn = fv * c[j] + iv * gv;
      c[j] = cn;
      h[j] = ov * fast_tanh(cn);
    }

    // ---- store h tile: lanes 0-15 of VGPR r write a contiguous 64B run ----
    float* outp = out + (size_t)t * BATCH * HID;
#pragma unroll
    for (int r = 0; r < 8; ++r) {
      const int M = r + 8 * hiHalf;
      outp[(size_t)(b0 + M) * HID + h0 + nIdx] = h[r];
    }
  }

  // ---- final h and c appended after outputs ----
  float* hout = out + (size_t)T_STEPS * BATCH * HID;
  float* cout = hout + (size_t)BATCH * HID;
#pragma unroll
  for (int r = 0; r < 8; ++r) {
    const int M = r + 8 * hiHalf;
    const size_t off = (size_t)(b0 + M) * HID + h0 + nIdx;
    hout[off] = h[r];
    cout[off] = c[r];
  }
}

// ---------------------------------------------------------------------------
extern "C" void kernel_launch(void* const* d_in, const int* in_sizes, int n_in,
                              void* d_out, int out_size, void* d_ws,
                              size_t ws_size, hipStream_t stream) {
  const float* inputs  = (const float*)d_in[0];   // [T,B,D]
  const float* centres = (const float*)d_in[1];   // [4,C,D+H]
  const float* lin_w   = (const float*)d_in[2];   // [4,H,C]
  const float* lin_b   = (const float*)d_in[3];   // [4,H]
  float* out = (float*)d_out;                     // outputs | h | c
  _Float16* kbuf = (_Float16*)d_ws;               // 4*65536*32 f16 = 16 MB

  qk_kernel<<<R_ROWS, 128, 0, stream>>>(inputs, centres, kbuf);

  const int tiles  = (BATCH / 16) * (HID / 16);   // 1024 waves
  const int blocks = tiles / 4;                   // 4 waves / block
  lstm_kernel<<<blocks, 128, 0, stream>>>(kbuf, lin_w, lin_b, out);
}
